// AdaptiveFractalFeedForward_43112881717572
// MI455X (gfx1250) — compile-verified, matrix operand order checked by reading, not run on hardware
//
#include <hip/hip_runtime.h>
#include <hip/hip_bf16.h>
#include <math.h>

typedef _Float16 f16;
typedef _Float16 v16h __attribute__((ext_vector_type(16)));
typedef _Float16 v8h  __attribute__((ext_vector_type(8)));
typedef float    v8f  __attribute__((ext_vector_type(8)));
typedef unsigned int u32x4 __attribute__((ext_vector_type(4)));
typedef int          i32x4 __attribute__((ext_vector_type(4)));
typedef int          i32x8 __attribute__((ext_vector_type(8)));

#define DIM   768
#define HID   3072
#define HID2  1536
#define NLEV  9
#define SEQ   2048
#define NTOK  4096
#define LDK   72              // LDS K-stride (halves): 64 + 8 pad (16B), rows 16B aligned
#define MAXTILES 272

#if defined(__has_builtin)
#if __has_builtin(__builtin_amdgcn_tensor_load_to_lds) && __has_builtin(__builtin_amdgcn_s_wait_tensorcnt)
#define HAVE_TDM 1
#endif
#endif

#define WMMA(a, b, c) __builtin_amdgcn_wmma_f32_16x16x32_f16(false, (a), false, (b), (short)0, (c), false, false)

// ---------------------------------------------------------------------------
// WMMA 16-bit operand fragment from LDS tile stored [row][k] (ISA 7.12.2):
// lanes 0-15 -> rows, lane bit4 selects K-half; two 16B chunks per lane.
// ---------------------------------------------------------------------------
__device__ __forceinline__ v16h load_frag(const f16* __restrict__ base, int ldk) {
    int lane = threadIdx.x & 31;
    int row  = lane & 15;
    int kh   = (lane >> 4) << 3;
    const f16* p = base + row * ldk + kh;
    v8h lo = *(const v8h*)(p);
    v8h hi = *(const v8h*)(p + 16);
    v16h r;
#pragma unroll
    for (int i = 0; i < 8; ++i) { r[i] = lo[i]; r[i + 8] = hi[i]; }
    return r;
}

#ifdef HAVE_TDM
// Low 32 bits of a flat shared-memory pointer == workgroup-relative LDS addr.
__device__ __forceinline__ unsigned int lds_of(const void* p) {
    return (unsigned int)(unsigned long long)p;
}
// 2-D TDM tile load: tile_rows x 64 halves, row stride = stride_elems, into
// LDS with +16B pad per 128B row (-> LDK=72). Tensor dims set huge (no OOB).
__device__ __forceinline__ void tdm_load_2d(unsigned int lds_addr, const void* gaddr,
                                            int tile_rows, int stride_elems) {
    unsigned long long ga = (unsigned long long)gaddr;
    u32x4 g0;
    g0[0] = 1u;                                            // count=1 (valid)
    g0[1] = lds_addr;
    g0[2] = (unsigned int)ga;
    g0[3] = (unsigned int)((ga >> 32) & 0x1FFFFFFu) | 0x80000000u;  // type=2
    i32x8 g1;
    g1[0] = (1 << 16) | (1 << 20) | (4 << 22) | (3 << 25); // 2B, pad_en, every 32dw +4dw
    g1[1] = 0;                                             // tdim0 lo16<<16 (2^30 -> 0)
    g1[2] = 0x4000;                                        // tdim0 hi16 ; tdim1 lo16
    g1[3] = 0x4000 | (64 << 16);                           // tdim1 hi16 ; tile_dim0=64
    g1[4] = tile_rows & 0xFFFF;                            // tile_dim1 ; tile_dim2=0
    g1[5] = stride_elems;                                  // dim0 stride lo32
    g1[6] = 0;
    g1[7] = 0;
    i32x4 z4 = {0, 0, 0, 0};
    i32x8 z8 = {0, 0, 0, 0, 0, 0, 0, 0};
    __builtin_amdgcn_tensor_load_to_lds(g0, g1, z4, z4, z8, 0);
}
// Gather-mode TDM load: 16 rows selected by 16-bit indices (groups 2/3),
// each row 64 halves from gaddr + idx*stride. Same LDS padding.
__device__ __forceinline__ void tdm_load_gather16(unsigned int lds_addr, const void* gaddr,
                                                  i32x4 g2, i32x4 g3, int stride_elems) {
    unsigned long long ga = (unsigned long long)gaddr;
    u32x4 g0;
    g0[0] = 1u | (1u << 31);                               // count=1, gather_mode, 16b idx
    g0[1] = lds_addr;
    g0[2] = (unsigned int)ga;
    g0[3] = (unsigned int)((ga >> 32) & 0x1FFFFFFu) | 0x80000000u;
    i32x8 g1;
    g1[0] = (1 << 16) | (1 << 20) | (4 << 22) | (3 << 25);
    g1[1] = 0;
    g1[2] = 0x4000;
    g1[3] = 0x4000 | (64 << 16);                           // tile_dim0 = 64
    g1[4] = 16;                                            // tile_dim1 = #indices
    g1[5] = stride_elems;
    g1[6] = 0;
    g1[7] = 0;
    i32x8 z8 = {0, 0, 0, 0, 0, 0, 0, 0};
    __builtin_amdgcn_tensor_load_to_lds(g0, g1, g2, g3, z8, 0);
}
#endif

// ---------------------------------------------------------------------------
// LayerNorm -> f16 (one token per block)
// ---------------------------------------------------------------------------
__global__ __launch_bounds__(256) void ln_kernel(const float* __restrict__ x,
                                                 const float* __restrict__ gamma,
                                                 const float* __restrict__ beta,
                                                 f16* __restrict__ xn) {
    __shared__ float s1[256], s2[256];
    size_t t = blockIdx.x;
    const float* xr = x + t * DIM;
    float a = 0.f, b = 0.f;
    for (int i = threadIdx.x; i < DIM; i += 256) { float v = xr[i]; a += v; b += v * v; }
    s1[threadIdx.x] = a; s2[threadIdx.x] = b; __syncthreads();
    for (int o = 128; o > 0; o >>= 1) {
        if (threadIdx.x < o) { s1[threadIdx.x] += s1[threadIdx.x + o]; s2[threadIdx.x] += s2[threadIdx.x + o]; }
        __syncthreads();
    }
    float mean = s1[0] * (1.f / DIM);
    float var  = s2[0] * (1.f / DIM) - mean * mean;
    float rs   = rsqrtf(var + 1e-5f);
    for (int i = threadIdx.x; i < DIM; i += 256)
        xn[t * DIM + i] = (f16)((xr[i] - mean) * rs * gamma[i] + beta[i]);
}

// ---------------------------------------------------------------------------
// f32 [R][C] -> f16 [C][R] transpose+convert (grid.z batches matrices)
// ---------------------------------------------------------------------------
__global__ __launch_bounds__(256) void transpose_cvt(const float* __restrict__ in,
                                                     f16* __restrict__ out,
                                                     int R, int C) {
    __shared__ f16 tile[32][33];
    size_t base = (size_t)blockIdx.z * R * C;
    int c = blockIdx.x * 32 + threadIdx.x;
#pragma unroll
    for (int j = 0; j < 4; ++j) {
        int r = blockIdx.y * 32 + threadIdx.y + j * 8;
        tile[threadIdx.y + j * 8][threadIdx.x] = (f16)in[base + (size_t)r * C + c];
    }
    __syncthreads();
#pragma unroll
    for (int j = 0; j < 4; ++j) {
        int ro = blockIdx.x * 32 + threadIdx.y + j * 8;
        int co = blockIdx.y * 32 + threadIdx.x;
        out[base + (size_t)ro * R + co] = tile[threadIdx.x][threadIdx.y + j * 8];
    }
}

// ---------------------------------------------------------------------------
// Routing: counting-sort tokens by level, 16-row tile metadata, softmax mix.
// ---------------------------------------------------------------------------
__global__ __launch_bounds__(256) void route_kernel(const int* __restrict__ levels_info,
                                                    const float* __restrict__ lmw,
                                                    int* __restrict__ perm,
                                                    float* __restrict__ mix,
                                                    int* __restrict__ tlevel,
                                                    int* __restrict__ tstart,
                                                    int* __restrict__ trows,
                                                    int* __restrict__ ntiles) {
    __shared__ int cnt[16];
    __shared__ int cursor[16];
    __shared__ float sumExp;
    int tid = threadIdx.x;
    if (tid < 16) cnt[tid] = 0;
    if (tid == 0) sumExp = 0.f;
    __syncthreads();
    float myExp = 0.f;
    for (int s = tid; s < SEQ; s += 256) {
        int l = levels_info[s * 4]; l = l < 0 ? 0 : (l > 8 ? 8 : l);
        atomicAdd(&cnt[l], 1);
        myExp += expf(lmw[l]);
    }
    atomicAdd(&sumExp, myExp);
    __syncthreads();
    if (tid == 0) {
        int off = 0, nt = 0;
        for (int l = 0; l < NLEV; ++l) {
            int c = cnt[l] * 2;
            cursor[l] = off;
            for (int i = 0; i < c; i += 16) {
                tlevel[nt] = l;
                tstart[nt] = off + i;
                trows[nt]  = (c - i) < 16 ? (c - i) : 16;
                ++nt;
            }
            off += c;
        }
        ntiles[0] = nt;
    }
    __syncthreads();
    for (int t = tid; t < NTOK; t += 256) {
        int s = t & (SEQ - 1);
        int l = levels_info[s * 4]; l = l < 0 ? 0 : (l > 8 ? 8 : l);
        int slot = atomicAdd(&cursor[l], 1);
        perm[slot] = t;
    }
    float denom = sumExp;
    for (int s = tid; s < SEQ; s += 256) {
        int l = levels_info[s * 4]; l = l < 0 ? 0 : (l > 8 ? 8 : l);
        mix[s] = expf(lmw[l]) / denom;
    }
}

// ---------------------------------------------------------------------------
// Main GEMM: C[M,N] = act(A[M,K] @ Bt[N,K]^T + bias).
// 128x128 block, 8 waves (4m x 2n), each wave 32x64 (8 WMMA / 32-K step).
// TDM double-buffered LDS staging (wave0 issues, all compute).
// ---------------------------------------------------------------------------
template <int ACT, typename OUT>
__global__ __launch_bounds__(256) void gemm_main(const f16* __restrict__ A,
                                                 const f16* __restrict__ Bt,
                                                 const float* __restrict__ bias,
                                                 OUT* __restrict__ C,
                                                 int M, int N, int K) {
    __shared__ __align__(16) f16 As[2][128 * LDK];
    __shared__ __align__(16) f16 Bs[2][128 * LDK];
    int tid = threadIdx.x, wid = tid >> 5, lane = tid & 31;
    int m0 = blockIdx.y * 128, n0 = blockIdx.x * 128;
    int wm = (wid & 3) * 32;
    int wn = (wid >> 2) * 64;
    v8f acc[2][4] = {};
    const int nst = K / 64;

    auto compute = [&](int cur) {
#pragma unroll
        for (int kk = 0; kk < 64; kk += 32) {
            v16h a0 = load_frag(&As[cur][(wm)      * LDK + kk], LDK);
            v16h a1 = load_frag(&As[cur][(wm + 16) * LDK + kk], LDK);
#pragma unroll
            for (int j = 0; j < 4; ++j) {
                v16h b = load_frag(&Bs[cur][(wn + j * 16) * LDK + kk], LDK);
                acc[0][j] = WMMA(a0, b, acc[0][j]);
                acc[1][j] = WMMA(a1, b, acc[1][j]);
            }
        }
    };

#ifdef HAVE_TDM
    if (wid == 0) {
        tdm_load_2d(lds_of(&As[0][0]), A  + (size_t)m0 * K, 128, K);
        tdm_load_2d(lds_of(&Bs[0][0]), Bt + (size_t)n0 * K, 128, K);
        __builtin_amdgcn_s_wait_tensorcnt(0);
    }
    __syncthreads();
    for (int i = 0; i < nst; ++i) {
        int cur = i & 1;
        if (wid == 0 && i + 1 < nst) {
            int k0 = (i + 1) * 64;
            tdm_load_2d(lds_of(&As[cur ^ 1][0]), A  + (size_t)m0 * K + k0, 128, K);
            tdm_load_2d(lds_of(&Bs[cur ^ 1][0]), Bt + (size_t)n0 * K + k0, 128, K);
        }
        compute(cur);
        __syncthreads();
        if (wid == 0) __builtin_amdgcn_s_wait_tensorcnt(0);
        __syncthreads();
    }
#else
    for (int i = 0; i < nst; ++i) {
        int k0 = i * 64;
        __syncthreads();
#pragma unroll
        for (int j = 0; j < 4; ++j) {
            int idx = tid + j * 256;
            int r = idx >> 3, c = (idx & 7) * 8;
            *(v8h*)&As[0][r * LDK + c] = *(const v8h*)&A [(size_t)(m0 + r) * K + k0 + c];
            *(v8h*)&Bs[0][r * LDK + c] = *(const v8h*)&Bt[(size_t)(n0 + r) * K + k0 + c];
        }
        __syncthreads();
        compute(0);
    }
#endif

    int nl = lane & 15, mh = (lane >> 4) * 8;
#pragma unroll
    for (int tm = 0; tm < 2; ++tm)
#pragma unroll
        for (int j = 0; j < 4; ++j) {
            int ng = n0 + wn + j * 16 + nl;
            float bv = bias[ng];
#pragma unroll
            for (int r = 0; r < 8; ++r) {
                int mg = m0 + wm + tm * 16 + mh + r;
                float v = acc[tm][j][r] + bv;
                if (ACT == 1) v = 0.5f * v * (1.0f + erff(v * 0.70710678118654752f));
                C[(size_t)mg * N + ng] = (OUT)v;
            }
        }
}

// ---------------------------------------------------------------------------
// Adapter GEMMs: 16-row level tiles x 128 cols, 8 waves, per-level weights.
// A rows gathered via TDM gather-mode (16-bit perm indices); B via 2-D TDM.
// PHASE 1: relu(xn[perm] @ A1t^T + a1b) -> HL (f16, by permuted slot)
// PHASE 2: HL @ A2t^T + a2b -> ADP (f32, scattered by perm)
// ---------------------------------------------------------------------------
template <int PHASE>
__global__ __launch_bounds__(256) void adapter_gemm(const f16* __restrict__ Ain,
                                                    const f16* __restrict__ Wt,
                                                    const float* __restrict__ bias,
                                                    const int* __restrict__ perm,
                                                    const int* __restrict__ tlevel,
                                                    const int* __restrict__ tstart,
                                                    const int* __restrict__ trows,
                                                    const int* __restrict__ ntiles,
                                                    void* __restrict__ outp) {
    if ((int)blockIdx.y >= ntiles[0]) return;
    constexpr int K = (PHASE == 1) ? DIM : HID2;
    constexpr int N = (PHASE == 1) ? HID2 : DIM;
    __shared__ __align__(16) f16 As[2][16 * LDK];
    __shared__ __align__(16) f16 Bs[2][128 * LDK];
    int lvl = tlevel[blockIdx.y], start = tstart[blockIdx.y], rows = trows[blockIdx.y];
    int n0 = blockIdx.x * 128;
    const f16* Bt = Wt + (size_t)lvl * N * K;
    int tid = threadIdx.x, wid = tid >> 5, lane = tid & 31;
    int wn = wid * 16;
    v8f acc = {};
    const int nst = K / 64;

    auto compute = [&](int cur) {
#pragma unroll
        for (int kk = 0; kk < 64; kk += 32) {
            v16h a = load_frag(&As[cur][kk], LDK);
            v16h b = load_frag(&Bs[cur][wn * LDK + kk], LDK);
            acc = WMMA(a, b, acc);
        }
    };

#ifdef HAVE_TDM
    // pack the 16 (padded) row indices for gather-mode TDM
    i32x4 g2, g3;
#pragma unroll
    for (int i = 0; i < 8; ++i) {
        int r0 = 2 * i,     rr0 = r0 < rows ? r0 : rows - 1;
        int r1 = 2 * i + 1, rr1 = r1 < rows ? r1 : rows - 1;
        int i0 = (PHASE == 1) ? perm[start + rr0] : (start + rr0);
        int i1 = (PHASE == 1) ? perm[start + rr1] : (start + rr1);
        int packed = (i0 & 0xFFFF) | (i1 << 16);
        if (i < 4) g2[i] = packed; else g3[i - 4] = packed;
    }
    if (wid == 0) {
        tdm_load_gather16(lds_of(&As[0][0]), Ain, g2, g3, K);
        tdm_load_2d(lds_of(&Bs[0][0]), Bt + (size_t)n0 * K, 128, K);
        __builtin_amdgcn_s_wait_tensorcnt(0);
    }
    __syncthreads();
    for (int i = 0; i < nst; ++i) {
        int cur = i & 1;
        if (wid == 0 && i + 1 < nst) {
            int k0 = (i + 1) * 64;
            tdm_load_gather16(lds_of(&As[cur ^ 1][0]), Ain + k0, g2, g3, K);
            tdm_load_2d(lds_of(&Bs[cur ^ 1][0]), Bt + (size_t)n0 * K + k0, 128, K);
        }
        compute(cur);
        __syncthreads();
        if (wid == 0) __builtin_amdgcn_s_wait_tensorcnt(0);
        __syncthreads();
    }
#else
    for (int i = 0; i < nst; ++i) {
        int k0 = i * 64;
        __syncthreads();
        if (tid < 128) {
            int r = tid >> 3, c = (tid & 7) * 8;
            int rr = r < rows ? r : rows - 1;
            int arow = (PHASE == 1) ? perm[start + rr] : (start + rr);
            *(v8h*)&As[0][r * LDK + c] = *(const v8h*)&Ain[(size_t)arow * K + k0 + c];
        }
#pragma unroll
        for (int j = 0; j < 4; ++j) {
            int idx = tid + j * 256;
            int r = idx >> 3, c = (idx & 7) * 8;
            *(v8h*)&Bs[0][r * LDK + c] = *(const v8h*)&Bt[(size_t)(n0 + r) * K + k0 + c];
        }
        __syncthreads();
        compute(0);
    }
#endif

    int nl = lane & 15, mh = (lane >> 4) * 8;
    int ng = n0 + wn + nl;
    float bv = bias[(size_t)lvl * N + ng];
#pragma unroll
    for (int r = 0; r < 8; ++r) {
        int m = mh + r;
        if (m < rows) {
            float v = acc[r] + bv;
            if (PHASE == 1) {
                v = v > 0.f ? v : 0.f;
                ((f16*)outp)[(size_t)(start + m) * N + ng] = (f16)v;
            } else {
                int tok = perm[start + m];
                ((float*)outp)[(size_t)tok * N + ng] = v;
            }
        }
    }
}

// ---------------------------------------------------------------------------
// Final blend: out = main*(1-mix) + adapter*mix  (in place)
// ---------------------------------------------------------------------------
__global__ __launch_bounds__(256) void mix_kernel(float* __restrict__ out,
                                                  const float* __restrict__ adp,
                                                  const float* __restrict__ mix,
                                                  int total) {
    int i = blockIdx.x * 256 + threadIdx.x;
    if (i >= total) return;
    int s = (i / DIM) & (SEQ - 1);
    float m = mix[s];
    out[i] = out[i] * (1.f - m) + adp[i] * m;
}

// ---------------------------------------------------------------------------
extern "C" void kernel_launch(void* const* d_in, const int* in_sizes, int n_in,
                              void* d_out, int out_size, void* d_ws, size_t ws_size,
                              hipStream_t stream) {
    const float* x      = (const float*)d_in[0];
    const int*   lvinfo = (const int*)  d_in[1];
    const float* gamma  = (const float*)d_in[2];
    const float* beta   = (const float*)d_in[3];
    const float* W1     = (const float*)d_in[4];
    const float* b1     = (const float*)d_in[5];
    const float* W2     = (const float*)d_in[6];
    const float* b2     = (const float*)d_in[7];
    const float* A1     = (const float*)d_in[8];
    const float* a1b    = (const float*)d_in[9];
    const float* A2     = (const float*)d_in[10];
    const float* a2b    = (const float*)d_in[11];
    const float* lmw    = (const float*)d_in[12];
    float* out = (float*)d_out;

    char* w = (char*)d_ws;
    auto carve = [&](size_t bytes) -> void* {
        void* p = (void*)w;
        w += (bytes + 255) & ~(size_t)255;
        return p;
    };
    f16*   XN   = (f16*)  carve((size_t)NTOK * DIM * 2);
    f16*   W1T  = (f16*)  carve((size_t)HID  * DIM * 2);
    f16*   W2T  = (f16*)  carve((size_t)DIM  * HID * 2);
    f16*   A1T  = (f16*)  carve((size_t)NLEV * HID2 * DIM * 2);
    f16*   A2T  = (f16*)  carve((size_t)NLEV * DIM * HID2 * 2);
    f16*   H    = (f16*)  carve((size_t)NTOK * HID * 2);
    f16*   HL   = (f16*)  carve((size_t)NTOK * HID2 * 2);
    float* ADP  = (float*)carve((size_t)NTOK * DIM * 4);
    int*   PERM = (int*)  carve((size_t)NTOK * 4);
    float* MIX  = (float*)carve((size_t)SEQ * 4);
    int*   TLV  = (int*)  carve(MAXTILES * 4);
    int*   TST  = (int*)  carve(MAXTILES * 4);
    int*   TRW  = (int*)  carve(MAXTILES * 4);
    int*   TNT  = (int*)  carve(16);

    dim3 tb(32, 8);
    transpose_cvt<<<dim3(HID / 32,  DIM / 32, 1),     tb, 0, stream>>>(W1, W1T, DIM,  HID);
    transpose_cvt<<<dim3(DIM / 32,  HID / 32, 1),     tb, 0, stream>>>(W2, W2T, HID,  DIM);
    transpose_cvt<<<dim3(HID2 / 32, DIM / 32, NLEV),  tb, 0, stream>>>(A1, A1T, DIM,  HID2);
    transpose_cvt<<<dim3(DIM / 32,  HID2 / 32, NLEV), tb, 0, stream>>>(A2, A2T, HID2, DIM);

    ln_kernel<<<NTOK, 256, 0, stream>>>(x, gamma, beta, XN);
    route_kernel<<<1, 256, 0, stream>>>(lvinfo, lmw, PERM, MIX, TLV, TST, TRW, TNT);

    gemm_main<1, f16>  <<<dim3(HID / 128, NTOK / 128), 256, 0, stream>>>(XN, W1T, b1, H,   NTOK, HID, DIM);
    gemm_main<0, float><<<dim3(DIM / 128, NTOK / 128), 256, 0, stream>>>(H,  W2T, b2, out, NTOK, DIM, HID);

    adapter_gemm<1><<<dim3(HID2 / 128, 265), 256, 0, stream>>>(XN, A1T, a1b, PERM, TLV, TST, TRW, TNT, HL);
    adapter_gemm<2><<<dim3(DIM  / 128, 265), 256, 0, stream>>>(HL, A2T, a2b, PERM, TLV, TST, TRW, TNT, ADP);

    mix_kernel<<<(NTOK * DIM + 255) / 256, 256, 0, stream>>>(out, ADP, MIX, NTOK * DIM);
}
